// FNOFactorizedPointCloud2D_7507602833610
// MI455X (gfx1250) — matrix operand order repaired and checked by reading, not compiled, not run on hardware
//
#include <hip/hip_runtime.h>
#include <hip/hip_bf16.h>

typedef __attribute__((ext_vector_type(16))) _Float16 v16h;
typedef __attribute__((ext_vector_type(8)))  float    v8f;

#define B_  4
#define N_  8192
#define W_  64
#define PI2 6.2831853071795864f
#define INV64 0.015625f

// ---------------------------------------------------------------------------
// CDNA5 async global->LDS copy (ASYNCcnt-tracked, VGPR-bypassing).
// 16 bytes per lane per issue. LDS dest offset = low 32 bits of flat address
// (ISA: LDS aperture maps addr[31:0] to the LDS offset).
// ---------------------------------------------------------------------------
__device__ __forceinline__ void async_copy_b128(const void* gptr, void* lptr) {
  unsigned lds          = (unsigned)(uintptr_t)lptr;
  unsigned long long ga = (unsigned long long)(uintptr_t)gptr;
  asm volatile("global_load_async_to_lds_b128 %0, %1, off"
               :: "v"(lds), "v"(ga) : "memory");
}
__device__ __forceinline__ void wait_async0() {
  asm volatile("s_wait_asynccnt 0x0" ::: "memory");
}

// ---------------------------------------------------------------------------
// Stage 1: pointwise prep. fc0 (2->64) to f16 A-matrix layout [b][c][n],
// plus separable NUFFT phase tables cx[b][n][32kx][re,im], cy[b][n][32ky][re,im]
// (cy has 31 valid ky, padded to 32). Only ~2M sincos total.
// ---------------------------------------------------------------------------
__global__ __launch_bounds__(256) void prep_kernel(
    const float* __restrict__ u, const float* __restrict__ fc0_w,
    const float* __restrict__ fc0_b,
    _Float16* __restrict__ uh, float* __restrict__ cx, float* __restrict__ cy)
{
  int idx = blockIdx.x * 256 + threadIdx.x;          // b*N + n
  int n = idx & (N_ - 1); int b = idx >> 13;
  float x0 = u[(size_t)idx * 2], x1 = u[(size_t)idx * 2 + 1];
  #pragma unroll 8
  for (int c = 0; c < W_; ++c) {
    float v = x0 * fc0_w[c] + x1 * fc0_w[W_ + c] + fc0_b[c];
    uh[((size_t)(b * W_ + c)) * N_ + n] = (_Float16)v;
  }
  float* cxp = cx + (size_t)idx * 64;
  #pragma unroll
  for (int kx = 0; kx < 32; ++kx) {
    int k1v = (kx < 16) ? kx : kx - 32;
    float s, c; __sincosf(-PI2 * x0 * (float)k1v, &s, &c);
    cxp[kx * 2] = c; cxp[kx * 2 + 1] = s;
  }
  float* cyp = cy + (size_t)idx * 64;
  #pragma unroll
  for (int ky = 0; ky < 32; ++ky) {
    float re = 0.f, im = 0.f;
    if (ky < 31) {
      int k2v = (ky < 16) ? ky : ky - 31;
      float s, c; __sincosf(-PI2 * x1 * (float)k2v, &s, &c);
      re = c; im = s;
    }
    cyp[ky * 2] = re; cyp[ky * 2 + 1] = im;
  }
}

// ---------------------------------------------------------------------------
// Stage 2: forward NUFFT as WMMA GEMM.
// Per batch: D[64c x 1024col] = uh[64 x 8192] * basis[8192 x 1024],
// col = (kx*16+ky)*2 + {re,im}. Block = 8 waves, covers 128 cols; wave w owns
// col-tile w and 4 row-tiles (4 accumulators). K loop: 256 steps of 32.
// A/cx/cy tiles staged with global_load_async_to_lds_b128; basis B tile
// rebuilt in LDS each step from the separable phases.
// ---------------------------------------------------------------------------
__global__ __launch_bounds__(256) void nufft_fwd_kernel(
    const _Float16* __restrict__ uh, const float* __restrict__ cx,
    const float* __restrict__ cy, float* __restrict__ uft)
{
  __shared__ alignas(16) _Float16 As[64 * 32];
  __shared__ alignas(16) _Float16 Bs[128 * 32];
  __shared__ alignas(16) float    Cxs[32 * 64];
  __shared__ alignas(16) float    Cys[32 * 64];

  const int b  = blockIdx.x >> 3;
  const int cb = blockIdx.x & 7;
  const int t  = threadIdx.x;
  const int lane = t & 31, wave = t >> 5;

  v8f acc[4];
  #pragma unroll
  for (int r = 0; r < 4; ++r)
    #pragma unroll
    for (int i = 0; i < 8; ++i) acc[r][i] = 0.0f;

  const int colB = t >> 1;             // 0..127 local basis column
  const int kset = (t & 1) * 16;
  const int gcol = cb * 128 + colB;
  const int mode = gcol >> 1;
  const int comp = gcol & 1;
  const int kx   = mode >> 4;
  const int ky   = mode & 15;
  const int arow = t >> 2, aj = (t & 3) * 8;

  for (int ks = 0; ks < 256; ++ks) {
    const int k0 = ks * 32;
    // stage A (uh tile) + cx/cy tiles: async global->LDS, no VGPR round-trip
    async_copy_b128(uh + ((size_t)(b * 64 + arow)) * N_ + k0 + aj,
                    As + arow * 32 + aj);
    {
      const float* sx = cx + ((size_t)b * N_ + k0) * 64 + t * 8;
      async_copy_b128(sx,     Cxs + t * 8);
      async_copy_b128(sx + 4, Cxs + t * 8 + 4);
      const float* sy = cy + ((size_t)b * N_ + k0) * 64 + t * 8;
      async_copy_b128(sy,     Cys + t * 8);
      async_copy_b128(sy + 4, Cys + t * 8 + 4);
    }
    if (ks + 1 < 256) {   // hint next K-tile -> global_prefetch_b8
      __builtin_prefetch(uh + ((size_t)(b * 64 + arow)) * N_ + k0 + 32 + aj, 0, 0);
      __builtin_prefetch(cx + ((size_t)b * N_ + k0 + 32) * 64 + t * 8, 0, 0);
    }
    wait_async0();          // own async copies landed in LDS
    __syncthreads();        // publish to all waves
    // build basis B tile [col][k] (K contiguous per column)
    {
      _Float16* bp = Bs + colB * 32 + kset;
      #pragma unroll
      for (int j = 0; j < 16; ++j) {
        const int nl = kset + j;
        const float xr = Cxs[nl * 64 + kx * 2], xi = Cxs[nl * 64 + kx * 2 + 1];
        const float yr = Cys[nl * 64 + ky * 2], yi = Cys[nl * 64 + ky * 2 + 1];
        const float v = comp ? (xr * yi + xi * yr) : (xr * yr - xi * yi);
        bp[j] = (_Float16)v;
      }
    }
    __syncthreads();
    // WMMA: B frag = 16 contiguous K per lane column; A frag = 16 contiguous K per lane row
    v16h bfrag;
    {
      const _Float16* p = Bs + (wave * 16 + (lane & 15)) * 32 + (lane >> 4) * 16;
      #pragma unroll
      for (int j = 0; j < 16; ++j) bfrag[j] = p[j];
    }
    const int arowL = lane & 15, ahalf = lane >> 4;
    #pragma unroll
    for (int rt = 0; rt < 4; ++rt) {
      v16h afrag;
      const _Float16* p0 = As + (rt * 16 + arowL) * 32 + ahalf * 16;
      #pragma unroll
      for (int j = 0; j < 16; ++j) afrag[j] = p0[j];
      acc[rt] = __builtin_amdgcn_wmma_f32_16x16x32_f16(
          false, afrag, false, bfrag, (short)0, acc[rt], false, false);
    }
    __syncthreads();
  }
  // epilogue (documented C/D layout: 8 rows per lane, lane-half M offset)
  {
    const int nl = lane & 15, mb = (lane >> 4) * 8;
    const int oc = cb * 128 + wave * 16 + nl;
    #pragma unroll
    for (int rt = 0; rt < 4; ++rt)
      #pragma unroll
      for (int r = 0; r < 8; ++r)
        uft[((size_t)(b * 64 + rt * 16 + mb + r)) * 1024 + oc] = acc[rt][r];
  }
}

// ---------------------------------------------------------------------------
// Stage 3a: full complex ifft over kx (32 nonzero rows, incl 1/64).
// acol[bc][s1][ky][re,im]
// ---------------------------------------------------------------------------
__global__ __launch_bounds__(256) void ifft_kx_kernel(
    const float* __restrict__ uft, float* __restrict__ acol)
{
  int idx = blockIdx.x * 256 + threadIdx.x;   // (bc, s1, ky)
  int ky = idx & 15; int s1 = (idx >> 4) & 63; int bc = idx >> 10;
  const float* f = uft + (size_t)bc * 1024 + ky * 2;
  float ar = 0.f, ai = 0.f;
  for (int kxm = 0; kxm < 32; ++kxm) {
    int k1v = (kxm < 16) ? kxm : kxm - 32;
    float s, c; __sincosf((PI2 * (1.f / 64.f)) * (float)(k1v * s1), &s, &c);
    float fr = f[kxm * 32], fi = f[kxm * 32 + 1];
    ar += fr * c - fi * s; ai += fr * s + fi * c;
  }
  float* o = acol + ((size_t)bc * 64 + s1) * 32 + ky * 2;
  o[0] = ar * INV64; o[1] = ai * INV64;
}

// ---------------------------------------------------------------------------
// Stage 3b: irfft over ky (Hermitian half-spectrum, DC imag ignored) + grid
// bias term bs0.  uc[b][c][s1][s2]
// ---------------------------------------------------------------------------
__global__ __launch_bounds__(256) void irfft_ky_grid_kernel(
    const float* __restrict__ acol, const float* __restrict__ bs0_w,
    const float* __restrict__ bs0_b, float* __restrict__ uc)
{
  int idx = blockIdx.x * 256 + threadIdx.x;   // (b,c,s1,s2)
  int s2 = idx & 63; int s1 = (idx >> 6) & 63; int c = (idx >> 12) & 63;
  const float* Ap = acol + ((size_t)(idx >> 12)) * 2048 + s1 * 32;
  float sum = Ap[0];
  #pragma unroll
  for (int ky = 1; ky < 16; ++ky) {
    float s, cc; __sincosf((PI2 * (1.f / 64.f)) * (float)(ky * s2), &s, &cc);
    sum += 2.f * (Ap[ky * 2] * cc - Ap[ky * 2 + 1] * s);
  }
  sum *= INV64;
  float gx = (float)s1 * (1.f / 63.f), gy = (float)s2 * (1.f / 63.f);
  sum += gx * bs0_w[c * 2] + gy * bs0_w[c * 2 + 1] + bs0_b[c];
  uc[idx] = sum;
}

// ---------------------------------------------------------------------------
// Stage 4a: rfft over s1 (only the 32 needed kx rows). pint[bc][kxm][s2][2]
// ---------------------------------------------------------------------------
__global__ __launch_bounds__(256) void rfft_kx_kernel(
    const float* __restrict__ uc, float* __restrict__ pint)
{
  int idx = blockIdx.x * 256 + threadIdx.x;   // (bc, kxm, s2)
  int s2 = idx & 63; int kxm = (idx >> 6) & 31; int bc = idx >> 11;
  int k1v = (kxm < 16) ? kxm : kxm - 32;
  float pr = 0.f, pi = 0.f;
  const float* base = uc + (size_t)bc * 4096 + s2;
  for (int s1 = 0; s1 < 64; ++s1) {
    float s, c; __sincosf(-(PI2 * (1.f / 64.f)) * (float)(k1v * s1), &s, &c);
    float v = base[s1 * 64];
    pr += v * c; pi += v * s;
  }
  float* o = pint + ((size_t)bc * 32 + kxm) * 128 + s2 * 2;
  o[0] = pr; o[1] = pi;
}

// ---------------------------------------------------------------------------
// Stage 4b: rfft over s2 (ky 0..15). ucft[bc][kxm*16+ky][2]
// ---------------------------------------------------------------------------
__global__ __launch_bounds__(256) void rfft_ky_kernel(
    const float* __restrict__ pint, float* __restrict__ ucft)
{
  int idx = blockIdx.x * 256 + threadIdx.x;   // (bc, kxm, ky)
  int ky = idx & 15; int kxm = (idx >> 4) & 31; int bc = idx >> 9;
  const float* p = pint + ((size_t)bc * 32 + kxm) * 128;
  float fr = 0.f, fi = 0.f;
  for (int s2 = 0; s2 < 64; ++s2) {
    float s, c; __sincosf(-(PI2 * (1.f / 64.f)) * (float)(ky * s2), &s, &c);
    float pr = p[s2 * 2], pim = p[s2 * 2 + 1];
    fr += pr * c - pim * s; fi += pr * s + pim * c;
  }
  float* o = ucft + ((size_t)bc * 512 + kxm * 16 + ky) * 2;
  o[0] = fr; o[1] = fi;
}

// ---------------------------------------------------------------------------
// Stage 5: spectral conv, per-mode complex 64x64 matvec (w1 rows 0..15,
// w2 rows 16..31). fout[b][o][kxm*16+ky][2]
// ---------------------------------------------------------------------------
__global__ __launch_bounds__(256) void specconv_kernel(
    const float* __restrict__ ucft,
    const float* __restrict__ w1_re, const float* __restrict__ w1_im,
    const float* __restrict__ w2_re, const float* __restrict__ w2_im,
    float* __restrict__ fout)
{
  int idx = blockIdx.x * 256 + threadIdx.x;   // (b,o,kxm,ky)
  int ky = idx & 15; int kxm = (idx >> 4) & 31; int o = (idx >> 9) & 63;
  int b = idx >> 15;
  const float *wre, *wim; int x;
  if (kxm < 16) { wre = w1_re; wim = w1_im; x = kxm; }
  else          { wre = w2_re; wim = w2_im; x = kxm - 16; }
  float fr = 0.f, fi = 0.f;
  for (int i = 0; i < 64; ++i) {
    const float* uf = ucft + ((size_t)((b * 64 + i) * 512) + kxm * 16 + ky) * 2;
    float ur = uf[0], ui = uf[1];
    size_t widx = (((size_t)(i * 64 + o)) * 16 + x) * 16 + ky;
    float wr = wre[widx], wi = wim[widx];
    fr += ur * wr - ui * wi;
    fi += ur * wi + ui * wr;
  }
  float* out = fout + ((size_t)(b * 64 + o) * 512 + kxm * 16 + ky) * 2;
  out[0] = fr; out[1] = fi;
}

// ---------------------------------------------------------------------------
// Stage 6-pre: pack inverse-NUFFT A matrix (f16, K = 32kx*31ky*{re,im}=1984),
// Hermitian extension for ky=16..30: conj(fout[31-kxm][31-ky]).
// ---------------------------------------------------------------------------
__global__ __launch_bounds__(256) void pack_inv_a_kernel(
    const float* __restrict__ fout, _Float16* __restrict__ apack)
{
  int idx = blockIdx.x * 256 + threadIdx.x;
  if (idx >= B_ * W_ * 992) return;
  int mode = idx % 992; int bc = idx / 992;
  int kxm = mode / 31; int ky = mode - kxm * 31;
  float ur, ui;
  if (ky < 16) {
    const float* p = fout + ((size_t)bc * 512 + kxm * 16 + ky) * 2;
    ur = p[0]; ui = p[1];
  } else {
    const float* p = fout + ((size_t)bc * 512 + (31 - kxm) * 16 + (31 - ky)) * 2;
    ur = p[0]; ui = -p[1];
  }
  apack[(size_t)bc * 1984 + mode * 2]     = (_Float16)ur;
  apack[(size_t)bc * 1984 + mode * 2 + 1] = (_Float16)ui;
}

// ---------------------------------------------------------------------------
// Stage 6: inverse NUFFT as WMMA GEMM.
// up[64c x 8192n] = Apack[64 x 1984] * basisT[1984 x 8192] per batch,
// basis row kappa=(mode,comp): Re/Im of cx*cy (conjugate phase folds so the
// real part of the inverse transform is Ur*Re + Ui*Im). 62 K-steps of 32.
// Block covers 128 points; wave w owns point-tile w and 4 row-tiles.
// A tile staged with async global->LDS.
// ---------------------------------------------------------------------------
__global__ __launch_bounds__(256) void nufft_inv_kernel(
    const _Float16* __restrict__ apack, const float* __restrict__ cx,
    const float* __restrict__ cy, float* __restrict__ up)
{
  __shared__ alignas(16) _Float16 As[64 * 32];
  __shared__ alignas(16) _Float16 Bs[128 * 32];

  const int b  = blockIdx.x >> 6;
  const int nb = blockIdx.x & 63;
  const int t  = threadIdx.x;
  const int lane = t & 31, wave = t >> 5;

  v8f acc[4];
  #pragma unroll
  for (int r = 0; r < 4; ++r)
    #pragma unroll
    for (int i = 0; i < 8; ++i) acc[r][i] = 0.0f;

  const int colB = t >> 1;
  const int kset = (t & 1) * 16;
  const int np   = nb * 128 + colB;
  const float* cxn = cx + ((size_t)b * N_ + np) * 64;
  const float* cyn = cy + ((size_t)b * N_ + np) * 64;
  const int arow = t >> 2, aj = (t & 3) * 8;

  for (int kt = 0; kt < 62; ++kt) {
    const int kk0 = kt * 32;
    async_copy_b128(apack + ((size_t)(b * 64 + arow)) * 1984 + kk0 + aj,
                    As + arow * 32 + aj);
    {
      _Float16* bp = Bs + colB * 32 + kset;
      #pragma unroll
      for (int j = 0; j < 16; ++j) {
        const int kap = kk0 + kset + j;
        const int md = kap >> 1, cm = kap & 1;
        const int kxm = md / 31;
        const int ky2 = md - kxm * 31;
        const float xr = cxn[kxm * 2], xi = cxn[kxm * 2 + 1];
        const float yr = cyn[ky2 * 2], yi = cyn[ky2 * 2 + 1];
        const float v = cm ? (xr * yi + xi * yr) : (xr * yr - xi * yi);
        bp[j] = (_Float16)v;
      }
    }
    wait_async0();
    __syncthreads();
    v16h bfrag;
    {
      const _Float16* p = Bs + (wave * 16 + (lane & 15)) * 32 + (lane >> 4) * 16;
      #pragma unroll
      for (int j = 0; j < 16; ++j) bfrag[j] = p[j];
    }
    const int arowL = lane & 15, ahalf = lane >> 4;
    #pragma unroll
    for (int rt = 0; rt < 4; ++rt) {
      v16h afrag;
      const _Float16* p0 = As + (rt * 16 + arowL) * 32 + ahalf * 16;
      #pragma unroll
      for (int j = 0; j < 16; ++j) afrag[j] = p0[j];
      acc[rt] = __builtin_amdgcn_wmma_f32_16x16x32_f16(
          false, afrag, false, bfrag, (short)0, acc[rt], false, false);
    }
    __syncthreads();
  }
  {
    const int nl = lane & 15, mb = (lane >> 4) * 8;
    const int gn = nb * 128 + wave * 16 + nl;
    #pragma unroll
    for (int rt = 0; rt < 4; ++rt)
      #pragma unroll
      for (int r = 0; r < 8; ++r)
        up[((size_t)(b * 64 + rt * 16 + mb + r)) * N_ + gn] = acc[rt][r];
  }
}

// ---------------------------------------------------------------------------
// Stage 7: head. up + bs1 term, fc1 + exact GELU, fc2.
// ---------------------------------------------------------------------------
__global__ __launch_bounds__(256) void head_kernel(
    const float* __restrict__ u, const float* __restrict__ up,
    const float* __restrict__ bs1_w, const float* __restrict__ bs1_b,
    const float* __restrict__ fc1_w, const float* __restrict__ fc1_b,
    const float* __restrict__ fc2_w, const float* __restrict__ fc2_b,
    float* __restrict__ out)
{
  int idx = blockIdx.x * 256 + threadIdx.x;   // b*N + n
  int n = idx & (N_ - 1); int b = idx >> 13;
  float x0 = u[(size_t)idx * 2], x1 = u[(size_t)idx * 2 + 1];
  float h[W_];
  #pragma unroll 8
  for (int c = 0; c < W_; ++c)
    h[c] = up[((size_t)(b * W_ + c)) * N_ + n] +
           x0 * bs1_w[c * 2] + x1 * bs1_w[c * 2 + 1] + bs1_b[c];
  float o = fc2_b[0];
  for (int j = 0; j < 128; ++j) {
    float a = fc1_b[j];
    #pragma unroll 16
    for (int c = 0; c < W_; ++c) a += h[c] * fc1_w[c * 128 + j];
    float g = 0.5f * a * (1.f + erff(a * 0.70710678118f));
    o += g * fc2_w[j];
  }
  out[idx] = o;
}

// ---------------------------------------------------------------------------
extern "C" void kernel_launch(void* const* d_in, const int* in_sizes, int n_in,
                              void* d_out, int out_size, void* d_ws, size_t ws_size,
                              hipStream_t stream)
{
  const float* u     = (const float*)d_in[0];
  const float* fc0_w = (const float*)d_in[1];
  const float* fc0_b = (const float*)d_in[2];
  const float* w1_re = (const float*)d_in[3];
  const float* w1_im = (const float*)d_in[4];
  const float* w2_re = (const float*)d_in[5];
  const float* w2_im = (const float*)d_in[6];
  const float* bs0_w = (const float*)d_in[7];
  const float* bs0_b = (const float*)d_in[8];
  const float* bs1_w = (const float*)d_in[9];
  const float* bs1_b = (const float*)d_in[10];
  const float* fc1_w = (const float*)d_in[11];
  const float* fc1_b = (const float*)d_in[12];
  const float* fc2_w = (const float*)d_in[13];
  const float* fc2_b = (const float*)d_in[14];
  float* out = (float*)d_out;

  char* base = (char*)d_ws;
  size_t off = 0;
  auto carve = [&](size_t bytes) -> void* {
    void* q = base + off;
    off += (bytes + 255) & ~(size_t)255;
    return q;
  };
  _Float16* uh    = (_Float16*)carve((size_t)B_ * W_ * N_ * 2);      //  4 MB
  float*    cx    = (float*)   carve((size_t)B_ * N_ * 64 * 4);      //  8 MB
  float*    cy    = (float*)   carve((size_t)B_ * N_ * 64 * 4);      //  8 MB
  float*    uft   = (float*)   carve((size_t)B_ * W_ * 1024 * 4);    //  1 MB
  float*    acol  = (float*)   carve((size_t)B_ * W_ * 2048 * 4);    //  2 MB
  float*    uc    = (float*)   carve((size_t)B_ * W_ * 4096 * 4);    //  4 MB
  float*    pint  = (float*)   carve((size_t)B_ * W_ * 4096 * 4);    //  4 MB
  float*    ucft  = (float*)   carve((size_t)B_ * W_ * 1024 * 4);    //  1 MB
  float*    fout  = (float*)   carve((size_t)B_ * W_ * 1024 * 4);    //  1 MB
  _Float16* apack = (_Float16*)carve((size_t)B_ * W_ * 1984 * 2);    //  1 MB
  float*    up    = (float*)   carve((size_t)B_ * W_ * N_ * 4);      //  8 MB

  prep_kernel<<<B_ * N_ / 256, 256, 0, stream>>>(u, fc0_w, fc0_b, uh, cx, cy);
  nufft_fwd_kernel<<<B_ * 8, 256, 0, stream>>>(uh, cx, cy, uft);
  ifft_kx_kernel<<<B_ * W_ * 64 * 16 / 256, 256, 0, stream>>>(uft, acol);
  irfft_ky_grid_kernel<<<B_ * W_ * 4096 / 256, 256, 0, stream>>>(acol, bs0_w, bs0_b, uc);
  rfft_kx_kernel<<<B_ * W_ * 32 * 64 / 256, 256, 0, stream>>>(uc, pint);
  rfft_ky_kernel<<<B_ * W_ * 32 * 16 / 256, 256, 0, stream>>>(pint, ucft);
  specconv_kernel<<<B_ * W_ * 32 * 16 / 256, 256, 0, stream>>>(
      ucft, w1_re, w1_im, w2_re, w2_im, fout);
  pack_inv_a_kernel<<<(B_ * W_ * 992 + 255) / 256, 256, 0, stream>>>(fout, apack);
  nufft_inv_kernel<<<B_ * 64, 256, 0, stream>>>(apack, cx, cy, up);
  head_kernel<<<B_ * N_ / 256, 256, 0, stream>>>(
      u, up, bs1_w, bs1_b, fc1_w, fc1_b, fc2_w, fc2_b, out);

  (void)in_sizes; (void)n_in; (void)out_size; (void)ws_size;
}
